// GraphAttentionMLP_20160576488084
// MI455X (gfx1250) — compile-verified, matrix operand order checked by reading, not run on hardware
//
#include <hip/hip_runtime.h>
#include <hip/hip_bf16.h>

// ---------------------------------------------------------------------------
// CDNA5 (gfx1250) fused graph-attention MLP.
// All GEMM stages use v_wmma_f32_16x16x32_bf16 (wave32 WMMA).
// Weights are pre-packed into bf16 fragment-major layout so every B-fragment
// is one 32B per-lane load; LDS-staged A operands are stored fragment-packed.
// ---------------------------------------------------------------------------

typedef __attribute__((ext_vector_type(16))) __bf16 v16bf;
typedef __attribute__((ext_vector_type(8)))  float  v8f;

#define DD  320
#define Hh  8
#define WNUMv 448
#define INV_SQRT3f 0.57735026918962576451f
#define INV_SQRT2f 0.70710678118654752440f

__device__ __forceinline__ v8f wmma_bf16(v16bf a, v16bf b, v8f c) {
  return __builtin_amdgcn_wmma_f32_16x16x32_bf16(false, a, false, b, (short)0, c, false, false);
}

// A-fragment (16x32 bf16) gathered from row-major f32 (row stride rs, col
// element stride cs).  ISA 7.12.2 A layout.
__device__ __forceinline__ v16bf ld_a_f32(const float* p, int rs, int cs, int lane) {
  int m = lane & 15;
  int kb = (lane & 16) ? 8 : 0;
  v16bf a;
#pragma unroll
  for (int e = 0; e < 16; ++e) {
    int k = kb + ((e < 8) ? e : (e + 8));
    a[e] = (__bf16)p[m * rs + k * cs];
  }
  return a;
}

// Packed fragment loads: one v16bf (32B) per lane.
__device__ __forceinline__ v16bf ld_b_pk(const __bf16* w, int Kt, int nt, int kt, int lane) {
  return ((const v16bf*)w)[(size_t)(nt * Kt + kt) * 32 + lane];
}
__device__ __forceinline__ v16bf ld_a_pk(const __bf16* p, int kt, int lane) {
  return ((const v16bf*)p)[kt * 32 + lane];
}

// inverse of the in-fragment K mapping: k' (0..31) -> (half, elem)
__device__ __forceinline__ void inv_k(int kk, int& half, int& e) {
  if (kk < 8)       { half = 0; e = kk; }
  else if (kk < 16) { half = 1; e = kk - 8; }
  else if (kk < 24) { half = 0; e = kk - 8; }
  else              { half = 1; e = kk - 16; }
}

__device__ __forceinline__ float sigf(float x) { return 1.0f / (1.0f + __expf(-x)); }
__device__ __forceinline__ float siluf(float x) { return x * sigf(x); }
__device__ __forceinline__ float sleakyf(float x) {
  return 0.6f * x + 0.4f * x * (2.0f * sigf(x) - 1.0f);
}

__device__ __forceinline__ unsigned enc_f(float f) {
  unsigned u = __float_as_uint(f);
  return (u & 0x80000000u) ? ~u : (u | 0x80000000u);
}
__device__ __forceinline__ float dec_f(unsigned u) {
  return (u & 0x80000000u) ? __uint_as_float(u & 0x7FFFFFFFu) : __uint_as_float(~u);
}

// ---------------------------------------------------------------------------
// Weight packer: row-major f32 K x N  ->  bf16 fragment-major [nt][kt][lane][16]
// ---------------------------------------------------------------------------
__global__ __launch_bounds__(256) void k_pack_b(const float* __restrict__ src,
                                                __bf16* __restrict__ dst,
                                                int K, int N) {
  int i = blockIdx.x * blockDim.x + threadIdx.x;
  if (i >= K * N) return;
  int k = i / N, n = i - k * N;
  int kt = k >> 5, kk = k & 31, nt = n >> 4, nn = n & 15;
  int half, e;
  inv_k(kk, half, e);
  int Kt = K >> 5;
  dst[((size_t)(nt * Kt + kt) * 32 + half * 16 + nn) * 16 + e] = (__bf16)src[i];
}

// ---------------------------------------------------------------------------
// Kernel 1: node-side linears.  16 nodes/wave, 5 waves/block.
// ---------------------------------------------------------------------------
__global__ __launch_bounds__(160) void k_node_linear(
    const float* __restrict__ ni, const __bf16* __restrict__ Wss,
    const __bf16* __restrict__ Wsv, const float* __restrict__ bs,
    const __bf16* __restrict__ Wds, const __bf16* __restrict__ Wdv,
    float* __restrict__ mss, float* __restrict__ mds,
    float* __restrict__ msv, float* __restrict__ mdv, int N) {
  int lane = threadIdx.x & 31;
  int wave = threadIdx.x >> 5;
  int tile = blockIdx.x * 5 + wave;
  int n0 = tile * 16;
  if (n0 >= N) return;
  const float* A = ni + (size_t)n0 * DD;
  int n = lane & 15, mb = (lane & 16) ? 8 : 0;

  v16bf as[4];
#pragma unroll
  for (int kt = 0; kt < 4; ++kt) as[kt] = ld_a_f32(A + kt * 32, DD, 1, lane);
#pragma unroll
  for (int nt = 0; nt < 8; ++nt) {
    v8f c0 = {}, c1 = {};
#pragma unroll
    for (int kt = 0; kt < 4; ++kt) {
      c0 = wmma_bf16(as[kt], ld_b_pk(Wss, 4, nt, kt, lane), c0);
      c1 = wmma_bf16(as[kt], ld_b_pk(Wds, 4, nt, kt, lane), c1);
    }
#pragma unroll
    for (int r = 0; r < 8; ++r) {
      size_t row = (size_t)(n0 + mb + r);
      mss[row * 128 + nt * 16 + n] = c0[r] + bs[nt * 16 + n];
      mds[row * 128 + nt * 16 + n] = c1[r];
    }
  }
  for (int c = 0; c < 3; ++c) {
    const float* Av = A + 128 + c;
    v16bf av[2];
#pragma unroll
    for (int kt = 0; kt < 2; ++kt) av[kt] = ld_a_f32(Av + kt * 32 * 3, DD, 3, lane);
#pragma unroll
    for (int nt = 0; nt < 4; ++nt) {
      v8f c0 = {}, c1 = {};
#pragma unroll
      for (int kt = 0; kt < 2; ++kt) {
        c0 = wmma_bf16(av[kt], ld_b_pk(Wsv, 2, nt, kt, lane), c0);
        c1 = wmma_bf16(av[kt], ld_b_pk(Wdv, 2, nt, kt, lane), c1);
      }
#pragma unroll
      for (int r = 0; r < 8; ++r) {
        size_t row = (size_t)(n0 + mb + r);
        int col = nt * 16 + n;
        msv[row * 192 + col * 3 + c] = c0[r];
        mdv[row * 192 + col * 3 + c] = c1[r];
      }
    }
  }
}

// ---------------------------------------------------------------------------
// Kernel 2: fused per-edge pipeline.  16 edges/wave, 2 waves/block.
// ---------------------------------------------------------------------------
#define EDGE_WAVES 2
#define PW_F32 20672                       // f32 staging floats per wave
#define PW_BYTES (PW_F32 * 4 + 8192 * 2)   // + packed bf16 (hb/ds/dvc)

__global__ __launch_bounds__(EDGE_WAVES * 32) void k_edge(
    const float* __restrict__ mss, const float* __restrict__ mds,
    const float* __restrict__ msv, const float* __restrict__ mdv,
    const int* __restrict__ esrc, const int* __restrict__ edst,
    const float* __restrict__ eattr, const float* __restrict__ escal,
    const __bf16* __restrict__ rad_W1, const float* __restrict__ rad_b1,
    const __bf16* __restrict__ rad_W2, const float* __restrict__ rad_off,
    const __bf16* __restrict__ Wact_s, const float* __restrict__ bact,
    const __bf16* __restrict__ Wact_v, const __bf16* __restrict__ Walpha,
    const float* __restrict__ balpha, const float* __restrict__ adot,
    const float* __restrict__ w_int, const __bf16* __restrict__ Wval_s,
    const float* __restrict__ bval, const __bf16* __restrict__ Wval_v,
    float* __restrict__ glogits, float* __restrict__ value, int E) {
  extern __shared__ char smem[];
  int lane = threadIdx.x & 31;
  int wave = threadIdx.x >> 5;
  int tile = blockIdx.x * EDGE_WAVES + wave;
  int e0 = tile * 16;
  if (e0 >= E) return;

  char* base = smem + (size_t)wave * PW_BYTES;
  float* sM   = (float*)base;          // 16x128 msg_s
  float* vM   = sM + 2048;             // 16x192 msg_v [e][k*3+c]
  float* wR   = vM + 3072;             // 16x448 radial weights
  float* t192 = wR + 7168;             // 16x192 scal scratch
  float* vals = t192 + 3072;           // 16x128 val_s
  float* gv   = vals + 2048;           // 16x192 gated_v -> val_v in place
  float* ea   = gv + 3072;             // 16x4 edge_attr
  float* lgt  = ea + 64;               // 16x8 logits
  __bf16* hbpk  = (__bf16*)(lgt + 128);  // packed radial hidden  (2 kt, 1024)
  __bf16* dspk  = hbpk + 1024;           // packed ds / vs        (6 kt, 3072)
  __bf16* dvcpk = dspk + 3072;           // packed dv channel     (8 kt, 4096)

  int n = lane & 15, mb = (lane & 16) ? 8 : 0;

  // ---- gather edge attr + messages (float4 vectorized) ------------------
  if (lane < 16) ((float4*)ea)[lane] = ((const float4*)eattr)[e0 + lane];
  for (int i = lane; i < 128; i += 32) lgt[i] = 0.0f;
  for (int i = lane; i < 512; i += 32) {          // 16 edges x 32 float4
    int e = i >> 5, q = i & 31;
    int es = esrc[e0 + e], ed = edst[e0 + e];
    float4 a = ((const float4*)mss)[(size_t)es * 32 + q];
    float4 b = ((const float4*)mds)[(size_t)ed * 32 + q];
    ((float4*)sM)[i] = make_float4(a.x + b.x, a.y + b.y, a.z + b.z, a.w + b.w);
  }
  for (int i = lane; i < 768; i += 32) {          // 16 edges x 48 float4
    int e = i / 48, q = i % 48;
    int es = esrc[e0 + e], ed = edst[e0 + e];
    float4 a = ((const float4*)msv)[(size_t)es * 48 + q];
    float4 b = ((const float4*)mdv)[(size_t)ed * 48 + q];
    ((float4*)vM)[i] = make_float4(a.x + b.x, a.y + b.y, a.z + b.z, a.w + b.w);
  }

  // ---- radial MLP: h = silu(es@W1+b1) -> packed; w = h@W2 + off ---------
  {
    v16bf a = ld_a_f32(escal + (size_t)e0 * 32, 32, 1, lane);
#pragma unroll
    for (int nt = 0; nt < 4; ++nt) {
      v8f c = {};
      c = wmma_bf16(a, ld_b_pk(rad_W1, 1, nt, 0, lane), c);
      int kt = nt >> 1;
      int half = n >> 3;
      int e = (nt & 1) * 8 + (n & 7);
#pragma unroll
      for (int r = 0; r < 8; ++r) {
        float h = siluf(c[r] + rad_b1[nt * 16 + n]);
        hbpk[((size_t)kt * 32 + half * 16 + mb + r) * 16 + e] = (__bf16)h;
      }
    }
  }
  {
    v16bf a0 = ld_a_pk(hbpk, 0, lane);
    v16bf a1 = ld_a_pk(hbpk, 1, lane);
    for (int nt = 0; nt < 28; ++nt) {
      v8f c = {};
      c = wmma_bf16(a0, ld_b_pk(rad_W2, 2, nt, 0, lane), c);
      c = wmma_bf16(a1, ld_b_pk(rad_W2, 2, nt, 1, lane), c);
#pragma unroll
      for (int r = 0; r < 8; ++r)
        wR[(mb + r) * WNUMv + nt * 16 + n] = c[r] + rad_off[nt * 16 + n];
    }
  }

  // ---- dtp #1 -> ds directly in packed-A layout -------------------------
  for (int p = lane; p < 3072; p += 32) {
    int kt = p >> 9, lp = (p >> 4) & 31, e = p & 15;
    int half = lp >> 4, row = lp & 15;
    int k2 = half ? ((e < 8) ? e + 8 : e + 16) : ((e < 8) ? e : e + 8);
    int m = kt * 32 + k2;
    float o;
    if (m < 128) {
      o = wR[row * WNUMv + m] * sM[row * 128 + m] * ea[row * 4];
    } else {
      int k = m - 128;
      float d = vM[row * 192 + k * 3] * ea[row * 4 + 1] +
                vM[row * 192 + k * 3 + 1] * ea[row * 4 + 2] +
                vM[row * 192 + k * 3 + 2] * ea[row * 4 + 3];
      o = wR[row * WNUMv + 320 + k] * d * INV_SQRT3f;
    }
    dspk[p] = (__bf16)o;
  }

  // ---- dv per channel (packed) -> gated_v = dv @ Wact_v -----------------
  for (int c = 0; c < 3; ++c) {
    int c1 = (c + 1) % 3, c2 = (c + 2) % 3;
    for (int p = lane; p < 4096; p += 32) {
      int kt = p >> 9, lp = (p >> 4) & 31, e = p & 15;
      int half = lp >> 4, row = lp & 15;
      int k2 = half ? ((e < 8) ? e + 8 : e + 16) : ((e < 8) ? e : e + 8);
      int m = kt * 32 + k2;
      float o;
      if (m < 128) {
        o = wR[row * WNUMv + 128 + m] * sM[row * 128 + m] * ea[row * 4 + 1 + c];
      } else if (m < 192) {
        int k = m - 128;
        o = wR[row * WNUMv + 256 + k] * vM[row * 192 + k * 3 + c] * ea[row * 4];
      } else {
        int k = m - 192;
        float cr = vM[row * 192 + k * 3 + c1] * ea[row * 4 + 1 + c2] -
                   vM[row * 192 + k * 3 + c2] * ea[row * 4 + 1 + c1];
        o = wR[row * WNUMv + 384 + k] * cr * INV_SQRT2f;
      }
      dvcpk[p] = (__bf16)o;
    }
    v16bf af[8];
#pragma unroll
    for (int kt = 0; kt < 8; ++kt) af[kt] = ld_a_pk(dvcpk, kt, lane);
#pragma unroll
    for (int nt = 0; nt < 4; ++nt) {
      v8f acc = {};
#pragma unroll
      for (int kt = 0; kt < 8; ++kt)
        acc = wmma_bf16(af[kt], ld_b_pk(Wact_v, 8, nt, kt, lane), acc);
#pragma unroll
      for (int r = 0; r < 8; ++r)
        gv[(mb + r) * 192 + (nt * 16 + n) * 3 + c] = acc[r];
    }
  }

  // ---- ds A-frags (shared by alpha + scal) ------------------------------
  v16bf ads[6];
#pragma unroll
  for (int kt = 0; kt < 6; ++kt) ads[kt] = ld_a_pk(dspk, kt, lane);

  // ---- attention logits -------------------------------------------------
  for (int h = 0; h < Hh; ++h) {
    v8f acc = {};
#pragma unroll
    for (int kt = 0; kt < 6; ++kt)
      acc = wmma_bf16(ads[kt], ld_b_pk(Walpha, 6, h, kt, lane), acc);
#pragma unroll
    for (int r = 0; r < 8; ++r) {
      float t = sleakyf(acc[r] + balpha[h * 16 + n]);
      float ctb = t * adot[h * 16 + n];
      ctb += __shfl_xor(ctb, 1, 32);
      ctb += __shfl_xor(ctb, 2, 32);
      ctb += __shfl_xor(ctb, 4, 32);
      ctb += __shfl_xor(ctb, 8, 32);
      if (n == 0) lgt[(mb + r) * 8 + h] = ctb;
    }
  }
  for (int i = lane; i < 128; i += 32) glogits[(size_t)e0 * 8 + i] = lgt[i];

  // ---- scal = ds @ Wact_s + bact ; val_s / val_v ------------------------
  for (int nt = 0; nt < 12; ++nt) {
    v8f acc = {};
#pragma unroll
    for (int kt = 0; kt < 6; ++kt)
      acc = wmma_bf16(ads[kt], ld_b_pk(Wact_s, 6, nt, kt, lane), acc);
#pragma unroll
    for (int r = 0; r < 8; ++r)
      t192[(mb + r) * 192 + nt * 16 + n] = acc[r] + bact[nt * 16 + n];
  }
  for (int i = lane; i < 2048; i += 32) {
    int e = i >> 7, m = i & 127;
    vals[i] = siluf(t192[e * 192 + m]);
  }
  for (int i = lane; i < 3072; i += 32) {
    int e = i / 192, k = (i % 192) / 3;
    gv[i] = gv[i] * sigf(t192[e * 192 + 128 + k]);
  }

  // ---- dtp #2 (scalar w_int): vs -> dspk --------------------------------
  for (int p = lane; p < 3072; p += 32) {
    int kt = p >> 9, lp = (p >> 4) & 31, e = p & 15;
    int half = lp >> 4, row = lp & 15;
    int k2 = half ? ((e < 8) ? e + 8 : e + 16) : ((e < 8) ? e : e + 8);
    int m = kt * 32 + k2;
    float o;
    if (m < 128) {
      o = w_int[m] * vals[row * 128 + m] * ea[row * 4];
    } else {
      int k = m - 128;
      float d = gv[row * 192 + k * 3] * ea[row * 4 + 1] +
                gv[row * 192 + k * 3 + 1] * ea[row * 4 + 2] +
                gv[row * 192 + k * 3 + 2] * ea[row * 4 + 3];
      o = w_int[320 + k] * d * INV_SQRT3f;
    }
    dspk[p] = (__bf16)o;
  }

  // ---- out_s = vs @ Wval_s + bval -> value[:, h, 0:16] ------------------
#pragma unroll
  for (int kt = 0; kt < 6; ++kt) ads[kt] = ld_a_pk(dspk, kt, lane);
  for (int nt = 0; nt < 8; ++nt) {
    v8f acc = {};
#pragma unroll
    for (int kt = 0; kt < 6; ++kt)
      acc = wmma_bf16(ads[kt], ld_b_pk(Wval_s, 6, nt, kt, lane), acc);
#pragma unroll
    for (int r = 0; r < 8; ++r) {
      int col = nt * 16 + n;
      value[((size_t)(e0 + mb + r)) * 320 + (col >> 4) * 40 + (col & 15)] =
          acc[r] + bval[col];
    }
  }

  // ---- vv channels -> out_v = vv @ Wval_v -> value[:, h, 16:40] ---------
  for (int c = 0; c < 3; ++c) {
    int c1 = (c + 1) % 3, c2 = (c + 2) % 3;
    for (int p = lane; p < 4096; p += 32) {
      int kt = p >> 9, lp = (p >> 4) & 31, e = p & 15;
      int half = lp >> 4, row = lp & 15;
      int k2 = half ? ((e < 8) ? e + 8 : e + 16) : ((e < 8) ? e : e + 8);
      int m = kt * 32 + k2;
      float o;
      if (m < 128) {
        o = w_int[128 + m] * vals[row * 128 + m] * ea[row * 4 + 1 + c];
      } else if (m < 192) {
        int k = m - 128;
        o = w_int[256 + k] * gv[row * 192 + k * 3 + c] * ea[row * 4];
      } else {
        int k = m - 192;
        float cr = gv[row * 192 + k * 3 + c1] * ea[row * 4 + 1 + c2] -
                   gv[row * 192 + k * 3 + c2] * ea[row * 4 + 1 + c1];
        o = w_int[384 + k] * cr * INV_SQRT2f;
      }
      dvcpk[p] = (__bf16)o;
    }
    v16bf af[8];
#pragma unroll
    for (int kt = 0; kt < 8; ++kt) af[kt] = ld_a_pk(dvcpk, kt, lane);
#pragma unroll
    for (int nt = 0; nt < 4; ++nt) {
      v8f acc = {};
#pragma unroll
      for (int kt = 0; kt < 8; ++kt)
        acc = wmma_bf16(af[kt], ld_b_pk(Wval_v, 8, nt, kt, lane), acc);
#pragma unroll
      for (int r = 0; r < 8; ++r) {
        int col = nt * 16 + n;  // 0..63
        value[((size_t)(e0 + mb + r)) * 320 + (col >> 3) * 40 + 16 +
              (col & 7) * 3 + c] = acc[r];
      }
    }
  }
}

// ---------------------------------------------------------------------------
// Segment softmax + aggregation
// ---------------------------------------------------------------------------
__global__ __launch_bounds__(256) void k_init(float* agg, float* den,
                                              unsigned* mx, int N) {
  int i = blockIdx.x * blockDim.x + threadIdx.x;
  if (i < N * 320) agg[i] = 0.0f;
  if (i < N * 8) { den[i] = 0.0f; mx[i] = 0u; }
}

__global__ __launch_bounds__(256) void k_max(const float* __restrict__ lg,
                                             const int* __restrict__ edst,
                                             unsigned* __restrict__ mx, int E) {
  int i = blockIdx.x * blockDim.x + threadIdx.x;
  if (i >= E * 8) return;
  int e = i >> 3, h = i & 7;
  atomicMax(&mx[(size_t)edst[e] * 8 + h], enc_f(lg[i]));
}

__global__ __launch_bounds__(256) void k_exp(float* __restrict__ lg,
                                             const int* __restrict__ edst,
                                             const unsigned* __restrict__ mx,
                                             float* __restrict__ den, int E) {
  int i = blockIdx.x * blockDim.x + threadIdx.x;
  if (i >= E * 8) return;
  int e = i >> 3, h = i & 7;
  unsigned u = mx[(size_t)edst[e] * 8 + h];
  float m = u ? dec_f(u) : 0.0f;
  float ex = __expf(lg[i] - m);
  lg[i] = ex;
  atomicAdd(&den[(size_t)edst[e] * 8 + h], ex);
}

__global__ __launch_bounds__(256) void k_agg(const float* __restrict__ ex,
                                             const float* __restrict__ den,
                                             const float* __restrict__ value,
                                             const int* __restrict__ edst,
                                             float* __restrict__ agg, int E) {
  int i = blockIdx.x * blockDim.x + threadIdx.x;
  if (i >= E * 320) return;
  int e = i / 320, r = i - e * 320, h = r / 40;
  int d = edst[e];
  float alpha = ex[(size_t)e * 8 + h] / (den[(size_t)d * 8 + h] + 1e-16f);
  atomicAdd(&agg[(size_t)d * 320 + r], value[(size_t)i] * alpha);
}

// ---------------------------------------------------------------------------
// Kernel 6: final node projection.  16 nodes/wave, 5 waves/block.
// ---------------------------------------------------------------------------
__global__ __launch_bounds__(160) void k_proj(const float* __restrict__ agg,
                                              const __bf16* __restrict__ Wps,
                                              const float* __restrict__ bp,
                                              const __bf16* __restrict__ Wpv,
                                              float* __restrict__ out, int N) {
  extern __shared__ char smem[];
  int lane = threadIdx.x & 31;
  int wave = threadIdx.x >> 5;
  int tile = blockIdx.x * 5 + wave;
  int n0 = tile * 16;
  if (n0 >= N) return;
  float* ns = (float*)(smem + (size_t)wave * (2048 + 3072) * 4);
  float* nv = ns + 2048;
  int n = lane & 15, mb = (lane & 16) ? 8 : 0;

  for (int i = lane; i < 2048; i += 32) {
    int e = i >> 7, m = i & 127;
    ns[i] = agg[(size_t)(n0 + e) * 320 + (m >> 4) * 40 + (m & 15)];
  }
  for (int i = lane; i < 3072; i += 32) {
    int e = i / 192, j = i % 192, k = j / 3, c = j % 3;
    nv[i] = agg[(size_t)(n0 + e) * 320 + (k >> 3) * 40 + 16 + (k & 7) * 3 + c];
  }

  v16bf as[4];
#pragma unroll
  for (int kt = 0; kt < 4; ++kt) as[kt] = ld_a_f32(ns + kt * 32, 128, 1, lane);
#pragma unroll
  for (int nt = 0; nt < 8; ++nt) {
    v8f acc = {};
#pragma unroll
    for (int kt = 0; kt < 4; ++kt)
      acc = wmma_bf16(as[kt], ld_b_pk(Wps, 4, nt, kt, lane), acc);
#pragma unroll
    for (int r = 0; r < 8; ++r)
      out[(size_t)(n0 + mb + r) * 320 + nt * 16 + n] = acc[r] + bp[nt * 16 + n];
  }
  for (int c = 0; c < 3; ++c) {
    v16bf av[2];
#pragma unroll
    for (int kt = 0; kt < 2; ++kt)
      av[kt] = ld_a_f32(nv + c + kt * 32 * 3, 192, 3, lane);
#pragma unroll
    for (int nt = 0; nt < 4; ++nt) {
      v8f acc = {};
#pragma unroll
      for (int kt = 0; kt < 2; ++kt)
        acc = wmma_bf16(av[kt], ld_b_pk(Wpv, 2, nt, kt, lane), acc);
#pragma unroll
      for (int r = 0; r < 8; ++r) {
        int col = nt * 16 + n;
        out[(size_t)(n0 + mb + r) * 320 + 128 + col * 3 + c] = acc[r];
      }
    }
  }
}

// ---------------------------------------------------------------------------
extern "C" void kernel_launch(void* const* d_in, const int* in_sizes, int n_in,
                              void* d_out, int out_size, void* d_ws, size_t ws_size,
                              hipStream_t stream) {
  (void)n_in; (void)out_size; (void)ws_size;
  const float* node_input   = (const float*)d_in[0];
  const int*   edge_src     = (const int*)d_in[1];
  const int*   edge_dst     = (const int*)d_in[2];
  const float* edge_attr    = (const float*)d_in[3];
  const float* edge_scalars = (const float*)d_in[4];
  const float* Wsrc_s = (const float*)d_in[5];
  const float* Wsrc_v = (const float*)d_in[6];
  const float* bsrc   = (const float*)d_in[7];
  const float* Wdst_s = (const float*)d_in[8];
  const float* Wdst_v = (const float*)d_in[9];
  const float* rad_W1 = (const float*)d_in[10];
  const float* rad_b1 = (const float*)d_in[11];
  const float* rad_W2 = (const float*)d_in[12];
  const float* rad_off = (const float*)d_in[13];
  const float* Wact_s = (const float*)d_in[14];
  const float* bact   = (const float*)d_in[15];
  const float* Wact_v = (const float*)d_in[16];
  const float* Walpha = (const float*)d_in[17];
  const float* balpha = (const float*)d_in[18];
  const float* w_int  = (const float*)d_in[19];
  const float* Wval_s = (const float*)d_in[20];
  const float* bval   = (const float*)d_in[21];
  const float* Wval_v = (const float*)d_in[22];
  const float* adot   = (const float*)d_in[23];
  const float* Wproj_s = (const float*)d_in[24];
  const float* bproj  = (const float*)d_in[25];
  const float* Wproj_v = (const float*)d_in[26];

  const int N = in_sizes[0] / DD;   // 10000
  const int E = in_sizes[1];        // 160000

  char* ws = (char*)d_ws;
  size_t off = 0;
  auto alloc = [&](size_t b) {
    size_t o = off;
    off += (b + 255) & ~(size_t)255;
    return o;
  };
  float*    mss   = (float*)(ws + alloc((size_t)N * 128 * 4));
  float*    mds   = (float*)(ws + alloc((size_t)N * 128 * 4));
  float*    msv   = (float*)(ws + alloc((size_t)N * 192 * 4));
  float*    mdv   = (float*)(ws + alloc((size_t)N * 192 * 4));
  float*    lgex  = (float*)(ws + alloc((size_t)E * 8 * 4));
  float*    value = (float*)(ws + alloc((size_t)E * 320 * 4));
  unsigned* mx    = (unsigned*)(ws + alloc((size_t)N * 8 * 4));
  float*    den   = (float*)(ws + alloc((size_t)N * 8 * 4));
  float*    agg   = (float*)(ws + alloc((size_t)N * 320 * 4));
  // packed bf16 weights
  __bf16* pWss = (__bf16*)(ws + alloc((size_t)128 * 128 * 2));
  __bf16* pWds = (__bf16*)(ws + alloc((size_t)128 * 128 * 2));
  __bf16* pWsv = (__bf16*)(ws + alloc((size_t)64 * 64 * 2));
  __bf16* pWdv = (__bf16*)(ws + alloc((size_t)64 * 64 * 2));
  __bf16* pRw1 = (__bf16*)(ws + alloc((size_t)32 * 64 * 2));
  __bf16* pRw2 = (__bf16*)(ws + alloc((size_t)64 * 448 * 2));
  __bf16* pWas = (__bf16*)(ws + alloc((size_t)192 * 192 * 2));
  __bf16* pWav = (__bf16*)(ws + alloc((size_t)256 * 64 * 2));
  __bf16* pWal = (__bf16*)(ws + alloc((size_t)192 * 128 * 2));
  __bf16* pWvs = (__bf16*)(ws + alloc((size_t)192 * 128 * 2));
  __bf16* pWvv = (__bf16*)(ws + alloc((size_t)256 * 64 * 2));
  __bf16* pWps = (__bf16*)(ws + alloc((size_t)128 * 128 * 2));
  __bf16* pWpv = (__bf16*)(ws + alloc((size_t)64 * 64 * 2));

  auto pack = [&](const float* s, __bf16* d, int K, int Nn) {
    k_pack_b<<<(K * Nn + 255) / 256, 256, 0, stream>>>(s, d, K, Nn);
  };
  pack(Wsrc_s, pWss, 128, 128);
  pack(Wdst_s, pWds, 128, 128);
  pack(Wsrc_v, pWsv, 64, 64);
  pack(Wdst_v, pWdv, 64, 64);
  pack(rad_W1, pRw1, 32, 64);
  pack(rad_W2, pRw2, 64, 448);
  pack(Wact_s, pWas, 192, 192);
  pack(Wact_v, pWav, 256, 64);
  pack(Walpha, pWal, 192, 128);
  pack(Wval_s, pWvs, 192, 128);
  pack(Wval_v, pWvv, 256, 64);
  pack(Wproj_s, pWps, 128, 128);
  pack(Wproj_v, pWpv, 64, 64);

  int node_tiles = (N + 15) / 16;
  int node_blocks = (node_tiles + 4) / 5;
  int edge_tiles = (E + 15) / 16;
  int edge_blocks = (edge_tiles + EDGE_WAVES - 1) / EDGE_WAVES;

  k_node_linear<<<node_blocks, 160, 0, stream>>>(
      node_input, pWss, pWsv, bsrc, pWds, pWdv, mss, mds, msv, mdv, N);

  k_init<<<(N * 320 + 255) / 256, 256, 0, stream>>>(agg, den, mx, N);

  k_edge<<<edge_blocks, EDGE_WAVES * 32, (size_t)EDGE_WAVES * PW_BYTES, stream>>>(
      mss, mds, msv, mdv, edge_src, edge_dst, edge_attr, edge_scalars,
      pRw1, rad_b1, pRw2, rad_off, pWas, bact, pWav, pWal, balpha,
      adot, w_int, pWvs, bval, pWvv, lgex, value, E);

  k_max<<<(E * 8 + 255) / 256, 256, 0, stream>>>(lgex, edge_dst, mx, E);
  k_exp<<<(E * 8 + 255) / 256, 256, 0, stream>>>(lgex, edge_dst, mx, den, E);
  k_agg<<<(E * 320 + 255) / 256, 256, 0, stream>>>(lgex, den, value, edge_dst, agg, E);

  k_proj<<<node_blocks, 160, (size_t)5 * (2048 + 3072) * 4, stream>>>(
      agg, pWps, bproj, pWpv, (float*)d_out, N);
}
